// AKT_44994077393137
// MI455X (gfx1250) — matrix operand order repaired and checked by reading, not compiled, not run on hardware
//
#include <hip/hip_runtime.h>

// ---------------------------------------------------------------------------
// AKT forward for MI455X (gfx1250, wave32). All GEMM-class math via
// V_WMMA_F32_16X16X32_BF16; softmax/LN/cumsum in fp32 VALU; LDS staging.
// GEMM uses register-staged double-buffered LDS so global load latency is
// hidden behind WMMA issue (one barrier per K-step, loads in flight during
// the 8-WMMA burst).
// ---------------------------------------------------------------------------

typedef __bf16 v16bf __attribute__((ext_vector_type(16)));
typedef float  v8f   __attribute__((ext_vector_type(8)));

#define BS 32
#define SEQ 512
#define T_TOK (BS * SEQ)   /* 16384 rows */
#define DM 512
#define NH 8
#define DK 64
#define DFF 2048
#define NQ 10000

// ---- device helpers -------------------------------------------------------

__device__ inline v8f vzero() {
    v8f z;
#pragma unroll
    for (int i = 0; i < 8; i++) z[i] = 0.f;
    return z;
}

__device__ inline v8f wmma_bf16(v16bf a, v16bf b, v8f c) {
    // D = A(16x32 bf16) * B(32x16 bf16) + C(16x16 f32)
    return __builtin_amdgcn_wmma_f32_16x16x32_bf16(false, a, false, b, (short)0, c,
                                                   false, false);
}

// Assemble a v16bf A/B operand from two contiguous 16-byte chunks.
union V16U { v16bf v; uint4 q[2]; };
__device__ inline v16bf ld2(const __bf16* p0, const __bf16* p1) {
    V16U u;
    u.q[0] = *(const uint4*)p0;
    u.q[1] = *(const uint4*)p1;
    return u.v;
}

__device__ inline float wave_sum(float v) {
#pragma unroll
    for (int m = 16; m > 0; m >>= 1) v += __shfl_xor(v, m, 32);
    return v;
}
__device__ inline float wave_max(float v) {
#pragma unroll
    for (int m = 16; m > 0; m >>= 1) v = fmaxf(v, __shfl_xor(v, m, 32));
    return v;
}

// ---- f32 -> bf16 weight conversion ----------------------------------------

__global__ __launch_bounds__(256) void convert_kernel(const float* __restrict__ src,
                                                      __bf16* __restrict__ dst, int n) {
    int i = blockIdx.x * 256 + threadIdx.x;
    if (i < n) dst[i] = (__bf16)src[i];
}

// ---- embedding + Rasch adjustment -----------------------------------------

__global__ __launch_bounds__(256) void embed_kernel(
    const float* __restrict__ q_embed, const float* __restrict__ qa_embed,
    const float* __restrict__ q_diff, const float* __restrict__ qa_diff,
    const float* __restrict__ diffp,
    const int* __restrict__ q_data, const int* __restrict__ target,
    const int* __restrict__ pid,
    float* __restrict__ xe_f, __bf16* __restrict__ xe_b,
    float* __restrict__ y_f, __bf16* __restrict__ y_b) {
    size_t idx = (size_t)blockIdx.x * 256 + threadIdx.x;
    int t = (int)(idx >> 9), d = (int)(idx & 511);
    int q = q_data[t], tg = target[t], p = pid[t];
    float pe = diffp[p];
    float qe = q_embed[(size_t)q * DM + d];
    float qd = q_diff[(size_t)q * DM + d];
    float qa = qa_embed[(size_t)tg * DM + d] + qe;
    float qad = qa_diff[(size_t)(q + NQ * tg) * DM + d];
    float xv = qe + pe * qd;
    float yv = qa + pe * (qad + qd);
    xe_f[idx] = xv; xe_b[idx] = (__bf16)xv;
    y_f[idx] = yv;  y_b[idx] = (__bf16)yv;
}

// ---- generic tiled bf16 WMMA GEMM: C[M,N] = A[M,K] * W[N,K]^T + bias ------
// Block: 256 threads = 8 waves; WG tile 128x128; wave tile 32x64; K step 32.
// Register-staged double buffering: tile k+1 global loads are in flight while
// the 8 WMMAs of tile k issue; single barrier per K-step.

__global__ __launch_bounds__(256) void gemm_bf16_kernel(
    const __bf16* __restrict__ A, const __bf16* __restrict__ W,
    const float* __restrict__ bias, float* __restrict__ Cf,
    __bf16* __restrict__ Cb, int M, int N, int K, int relu) {
    __shared__ __bf16 As[2][128 * 32];
    __shared__ __bf16 Bs[2][128 * 32];
    int tid = threadIdx.x;
    int m0 = blockIdx.y * 128, n0 = blockIdx.x * 128;
    int r = tid >> 1, hs = tid & 1;               // staging: 2 threads / row
    int w = tid >> 5, lane = tid & 31;
    int wm = w >> 1, wn = w & 1;                  // 4x2 wave grid
    int ml = lane & 15;
    bool hi = lane >= 16;
    v8f acc[2][4];
#pragma unroll
    for (int i = 0; i < 2; i++)
#pragma unroll
        for (int j = 0; j < 4; j++) acc[i][j] = vzero();

    const __bf16* arow = &A[(size_t)(m0 + r) * K + hs * 16];
    const __bf16* wrow = &W[(size_t)(n0 + r) * K + hs * 16];

    // prologue: stage tile 0 into registers
    uint4 ra = *(const uint4*)arow;
    uint4 rb = *(const uint4*)wrow;

    int cur = 0;
    for (int kb = 0; kb < K; kb += 32, cur ^= 1) {
        // commit staged registers to the current LDS buffer
        *(uint4*)&As[cur][r * 32 + hs * 16] = ra;
        *(uint4*)&Bs[cur][r * 32 + hs * 16] = rb;
        __syncthreads();
        // issue next tile's global loads now; they complete while we WMMA
        if (kb + 32 < K) {
            ra = *(const uint4*)(arow + kb + 32);
            rb = *(const uint4*)(wrow + kb + 32);
            if (kb + 64 < K) {   // warm WGP cache for the tile after next
                __builtin_prefetch(arow + kb + 64, 0, 3);
                __builtin_prefetch(wrow + kb + 64, 0, 3);
            }
        }
        v16bf a[2], bb[4];
#pragma unroll
        for (int i = 0; i < 2; i++) {
            // A operand (16x32): lane<16 holds K {0..7,16..23}, hi lanes {8..15,24..31}
            const __bf16* p = &As[cur][(wm * 32 + i * 16 + ml) * 32 + (hi ? 8 : 0)];
            a[i] = ld2(p, p + 16);
        }
#pragma unroll
        for (int j = 0; j < 4; j++) {
            // B operand (32x16): lane<16 holds K {0..7,16..23} of column n, hi lanes {8..15,24..31}
            const __bf16* p = &Bs[cur][(wn * 64 + j * 16 + ml) * 32 + (hi ? 8 : 0)];
            bb[j] = ld2(p, p + 16);
        }
#pragma unroll
        for (int i = 0; i < 2; i++)
#pragma unroll
            for (int j = 0; j < 4; j++) acc[i][j] = wmma_bf16(a[i], bb[j], acc[i][j]);
        // no trailing barrier: the buffer written next iteration is the other
        // one; the buffer we just read is only overwritten after the next
        // barrier, which every wave reaches only after finishing these reads.
    }
#pragma unroll
    for (int i = 0; i < 2; i++)
#pragma unroll
        for (int j = 0; j < 4; j++)
#pragma unroll
            for (int rr = 0; rr < 8; rr++) {
                int m = m0 + wm * 32 + i * 16 + rr + (hi ? 8 : 0);
                int n = n0 + wn * 64 + j * 16 + ml;
                float v = acc[i][j][rr] + bias[n];
                if (relu) v = fmaxf(v, 0.f);
                if (Cf) Cf[(size_t)m * N + n] = v;
                if (Cb) Cb[(size_t)m * N + n] = (__bf16)v;
            }
}

// ---- fused AKT attention per (batch, head, 64-row tile) -------------------
// 128 threads = 4 waves; each wave owns 16 query rows.
// Dynamic LDS: Kt[512][64] bf16 (64KB) + Vt[64][512] bf16 (64KB)
//            + per-wave 16x512 f32 score strip (4 x 32KB) = 256KB.

__global__ __launch_bounds__(128) void akt_attention_kernel(
    const __bf16* __restrict__ qk, const __bf16* __restrict__ vv,
    const float* __restrict__ gammas, __bf16* __restrict__ o,
    int mask_k, int zero_pad) {
    extern __shared__ char smem[];
    __bf16* Kt = (__bf16*)smem;                    // [key][d]
    __bf16* Vt = (__bf16*)(smem + 65536);          // [d][key] (transposed)
    int tid = threadIdx.x;
    int w = tid >> 5, lane = tid & 31;
    float* Sw = (float*)(smem + 131072 + w * 32768);  // [16][512] per wave
    int qt = blockIdx.x & 7;
    int bh = blockIdx.x >> 3;
    int h = bh & 7, b = bh >> 3;

    // stage K tile (this (b,h) slice of the shared Q/K projection)
    for (int idx = tid; idx < 512 * 8; idx += 128) {
        int row = idx >> 3, c = (idx & 7) * 8;
        *(uint4*)&Kt[row * 64 + c] =
            *(const uint4*)&qk[((size_t)(b * SEQ + row)) * DM + h * DK + c];
    }
    // stage V tile transposed
    for (int idx = tid; idx < 512 * 8; idx += 128) {
        int key = idx >> 3, dc = (idx & 7) * 8;
        union { uint4 q; __bf16 hh[8]; } u;
        u.q = *(const uint4*)&vv[((size_t)(b * SEQ + key)) * DM + h * DK + dc];
#pragma unroll
        for (int e = 0; e < 8; e++) Vt[(dc + e) * 512 + key] = u.hh[e];
    }
    __syncthreads();

    int ml = lane & 15;
    bool hi = lane >= 16;
    int rowbase = qt * 64 + w * 16;

    // Q operands for this wave's 16 rows (held across the whole N loop)
    const __bf16* qrow =
        qk + ((size_t)(b * SEQ + rowbase + ml)) * DM + h * DK + (hi ? 8 : 0);
    v16bf aq0 = ld2(qrow, qrow + 16);        // d 0..31
    v16bf aq1 = ld2(qrow + 32, qrow + 48);   // d 32..63

    // scores = Q * K^T / sqrt(dk)
    for (int nt = 0; nt < 32; nt++) {
        v8f acc = vzero();
        const __bf16* kr0 = &Kt[(nt * 16 + ml) * 64 + (hi ? 8 : 0)];
        acc = wmma_bf16(aq0, ld2(kr0, kr0 + 16), acc);
        const __bf16* kr1 = kr0 + 32;
        acc = wmma_bf16(aq1, ld2(kr1, kr1 + 16), acc);
#pragma unroll
        for (int rr = 0; rr < 8; rr++) {
            int m = rr + (hi ? 8 : 0);
            Sw[m * 512 + nt * 16 + ml] = acc[rr] * 0.125f;  // 1/sqrt(64)
        }
    }

    float gam = -logf(1.f + expf(gammas[h]));  // -softplus
    __bf16* Sb = (__bf16*)Sw;                   // bf16 reuse of score strip

    for (int m = 0; m < 16; m++) {
        int i = rowbase + m;
        int j0 = lane * 16;
        float raw[16], al[16];
#pragma unroll
        for (int t = 0; t < 16; t++) {
            raw[t] = Sw[m * 512 + j0 + t];
            al[t] = ((j0 + t) - i < mask_k) ? 1.f : 0.f;
        }
        // softmax #1 with -1e32 masking, then *maskf
        float mx = -1e32f;
#pragma unroll
        for (int t = 0; t < 16; t++) mx = fmaxf(mx, al[t] != 0.f ? raw[t] : -1e32f);
        mx = wave_max(mx);
        float p[16], s = 0.f;
#pragma unroll
        for (int t = 0; t < 16; t++) {
            p[t] = __expf((al[t] != 0.f ? raw[t] : -1e32f) - mx);
            s += p[t];
        }
        s = wave_sum(s);
        float inv = 1.f / s;
        // masked-softmax cumsum along k (lane-local + shfl prefix scan)
        float run = 0.f, cum[16];
#pragma unroll
        for (int t = 0; t < 16; t++) {
            float sc = p[t] * inv * al[t];
            run += sc;
            cum[t] = run;
        }
        float pre = run;
        for (int d = 1; d < 32; d <<= 1) {
            float o2 = __shfl_up(pre, d);
            if (lane >= d) pre += o2;
        }
        float excl = pre - run;
        float tot = __shfl(pre, 31);
        // distance decay + softmax #2
        float v2[16];
#pragma unroll
        for (int t = 0; t < 16; t++) {
            float dc = excl + cum[t];
            float pos = fabsf((float)(i - (j0 + t)));
            float ds = sqrtf(fmaxf((tot - dc) * pos, 0.f));
            float te = fminf(fmaxf(__expf(ds * gam), 1e-5f), 1e5f);
            v2[t] = al[t] != 0.f ? raw[t] * te : -1e32f;
        }
        float mx2 = -1e32f;
#pragma unroll
        for (int t = 0; t < 16; t++) mx2 = fmaxf(mx2, v2[t]);
        mx2 = wave_max(mx2);
        float e2[16], s2 = 0.f;
#pragma unroll
        for (int t = 0; t < 16; t++) {
            e2[t] = __expf(v2[t] - mx2);
            s2 += e2[t];
        }
        s2 = wave_sum(s2);
        float inv2 = 1.f / s2;
        bool zp = zero_pad && (i == 0);
#pragma unroll
        for (int t = 0; t < 16; t++) {
            float fv = zp ? 0.f : e2[t] * inv2;
            Sb[m * 512 + j0 + t] = (__bf16)fv;   // safe: stays below f32 row m+1
        }
    }

    // out = scores2 * V  (M=16, K=512, N=64)
    v8f accd[4];
#pragma unroll
    for (int nc = 0; nc < 4; nc++) accd[nc] = vzero();
    for (int kk = 0; kk < 16; kk++) {
        int kb = kk * 32;
        const __bf16* ar = &Sb[ml * 512 + kb + (hi ? 8 : 0)];
        v16bf a = ld2(ar, ar + 16);
#pragma unroll
        for (int nc = 0; nc < 4; nc++) {
            const __bf16* br = &Vt[(nc * 16 + ml) * 512 + kb + (hi ? 8 : 0)];
            accd[nc] = wmma_bf16(a, ld2(br, br + 16), accd[nc]);
        }
    }
#pragma unroll
    for (int nc = 0; nc < 4; nc++)
#pragma unroll
        for (int rr = 0; rr < 8; rr++) {
            int m = rr + (hi ? 8 : 0);
            int row = b * SEQ + rowbase + m;
            int col = h * DK + nc * 16 + ml;
            o[(size_t)row * DM + col] = (__bf16)accd[nc][rr];
        }
}

// ---- residual + LayerNorm (one wave per 512-wide row) ---------------------

__global__ __launch_bounds__(256) void ln_residual_kernel(
    float* __restrict__ X, const float* __restrict__ R,
    const float* __restrict__ g, const float* __restrict__ bta,
    __bf16* __restrict__ Yb) {
    int row = blockIdx.x * 8 + (threadIdx.x >> 5);
    int lane = threadIdx.x & 31;
    float v[16], s = 0.f, ss = 0.f;
#pragma unroll
    for (int t = 0; t < 16; t++) {
        int d = lane + t * 32;
        float x = X[(size_t)row * DM + d] + R[(size_t)row * DM + d];
        v[t] = x; s += x; ss += x * x;
    }
    s = wave_sum(s); ss = wave_sum(ss);
    float mu = s * (1.f / DM);
    float var = ss * (1.f / DM) - mu * mu;
    float rs = rsqrtf(var + 1e-5f);
#pragma unroll
    for (int t = 0; t < 16; t++) {
        int d = lane + t * 32;
        float y = (v[t] - mu) * rs * g[d] + bta[d];
        X[(size_t)row * DM + d] = y;
        Yb[(size_t)row * DM + d] = (__bf16)y;
    }
}

// ---- concat [x, q_embed_data] ---------------------------------------------

__global__ __launch_bounds__(256) void concat_kernel(const __bf16* __restrict__ xb,
                                                     const __bf16* __restrict__ qb,
                                                     __bf16* __restrict__ dst) {
    size_t idx = (size_t)blockIdx.x * 256 + threadIdx.x;
    int t = (int)(idx >> 10), c = (int)(idx & 1023);
    dst[idx] = (c < 512) ? xb[(size_t)t * DM + c] : qb[(size_t)t * DM + (c - 512)];
}

// ---- final N=1 projection + sigmoid ---------------------------------------

__global__ __launch_bounds__(256) void out3_kernel(const float* __restrict__ H,
                                                   const float* __restrict__ w3,
                                                   const float* __restrict__ b3,
                                                   float* __restrict__ out) {
    int row = blockIdx.x * 8 + (threadIdx.x >> 5);
    int lane = threadIdx.x & 31;
    float s = 0.f;
#pragma unroll
    for (int t = 0; t < 8; t++) {
        int c = lane + t * 32;
        s += H[(size_t)row * 256 + c] * w3[c];
    }
    s = wave_sum(s);
    if (lane == 0) out[row] = 1.f / (1.f + __expf(-(s + b3[0])));
}

// ---------------------------------------------------------------------------
// Host orchestration.
// Input order = setup_inputs() dict insertion order, flattened depth-first:
//   0 q_embed  1 qa_embed  2 q_embed_diff  3 qa_embed_diff  4 difficult_param
//   5 + 15*L for layer L in [blocks_1[0..1], blocks_2[0..3]]:
//     +0 k.w +1 k.b +2 v.w +3 v.b +4 out.w +5 out.b +6 gammas
//     +7 ln1.g +8 ln1.b +9 lin1.w +10 lin1.b +11 lin2.w +12 lin2.b +13 ln2.g +14 ln2.b
//   95 out1.w  96 out1.b  97 out2.w  98 out2.b  99 out3.w  100 out3.b
//   101 q_data  102 target  103 pid_data
// ---------------------------------------------------------------------------

static inline int LBASE(int L) { return 5 + 15 * L; }

extern "C" void kernel_launch(void* const* d_in, const int* in_sizes, int n_in,
                              void* d_out, int out_size, void* d_ws, size_t ws_size,
                              hipStream_t stream) {
    (void)in_sizes; (void)n_in; (void)out_size; (void)ws_size;

    char* base = (char*)d_ws;
    size_t off = 0;
    auto alloc = [&](size_t bytes) -> void* {
        void* p = base + off;
        off = (off + bytes + 255) & ~(size_t)255;
        return p;
    };
    const size_t TD = (size_t)T_TOK * DM;

    float*  qe_f = (float*)alloc(TD * 4);
    __bf16* qe_b = (__bf16*)alloc(TD * 2);
    float*  y_f  = (float*)alloc(TD * 4);
    __bf16* y_b  = (__bf16*)alloc(TD * 2);
    float*  x_f  = (float*)alloc(TD * 4);
    __bf16* x_b  = (__bf16*)alloc(TD * 2);
    __bf16* qk_b = (__bf16*)alloc(TD * 2);
    __bf16* v_b  = (__bf16*)alloc(TD * 2);
    __bf16* o_b  = (__bf16*)alloc(TD * 2);
    float*  tmp_f = (float*)alloc(TD * 4);                   // attn-out / ffn2 / fc2
    __bf16* h_b  = (__bf16*)alloc((size_t)T_TOK * DFF * 2);  // ffn hidden / concat
    __bf16 *wkb[6], *wvb[6], *wob[6], *w1b[6], *w2b[6];
    for (int L = 0; L < 6; L++) {
        wkb[L] = (__bf16*)alloc((size_t)DM * DM * 2);
        wvb[L] = (__bf16*)alloc((size_t)DM * DM * 2);
        wob[L] = (__bf16*)alloc((size_t)DM * DM * 2);
        w1b[L] = (__bf16*)alloc((size_t)DFF * DM * 2);
        w2b[L] = (__bf16*)alloc((size_t)DM * DFF * 2);
    }
    __bf16* o1wb = (__bf16*)alloc((size_t)512 * 1024 * 2);
    __bf16* o2wb = (__bf16*)alloc((size_t)256 * 512 * 2);

    auto conv = [&](int idx, __bf16* dst, size_t n) {
        convert_kernel<<<(unsigned)((n + 255) / 256), 256, 0, stream>>>(
            (const float*)d_in[idx], dst, (int)n);
    };
    for (int L = 0; L < 6; L++) {
        int B0 = LBASE(L);
        conv(B0 + 0, wkb[L], (size_t)DM * DM);
        conv(B0 + 2, wvb[L], (size_t)DM * DM);
        conv(B0 + 4, wob[L], (size_t)DM * DM);
        conv(B0 + 9, w1b[L], (size_t)DFF * DM);
        conv(B0 + 11, w2b[L], (size_t)DM * DFF);
    }
    conv(95, o1wb, (size_t)512 * 1024);
    conv(97, o2wb, (size_t)256 * 512);

    auto gemm = [&](const __bf16* Ap, const __bf16* Wp, const float* biasp,
                    float* Cf, __bf16* Cb, int M, int N, int K, int relu) {
        dim3 g(N / 128, M / 128);
        gemm_bf16_kernel<<<g, 256, 0, stream>>>(Ap, Wp, biasp, Cf, Cb, M, N, K, relu);
    };

    auto run_layer = [&](int L, int mask_k, int apply_pos, float* cx_f, __bf16* cx_b,
                         const __bf16* vsrc) {
        int B0 = LBASE(L);
        // shared Q/K projection (kq_same: q-input == k-input in every call)
        gemm(cx_b, wkb[L], (const float*)d_in[B0 + 1], nullptr, qk_b, T_TOK, DM, DM, 0);
        gemm(vsrc, wvb[L], (const float*)d_in[B0 + 3], nullptr, v_b, T_TOK, DM, DM, 0);
        akt_attention_kernel<<<BS * NH * (SEQ / 64), 128, 262144, stream>>>(
            qk_b, v_b, (const float*)d_in[B0 + 6], o_b, mask_k, mask_k == 0 ? 1 : 0);
        gemm(o_b, wob[L], (const float*)d_in[B0 + 5], tmp_f, nullptr, T_TOK, DM, DM, 0);
        ln_residual_kernel<<<T_TOK / 8, 256, 0, stream>>>(
            cx_f, tmp_f, (const float*)d_in[B0 + 7], (const float*)d_in[B0 + 8], cx_b);
        if (apply_pos) {
            gemm(cx_b, w1b[L], (const float*)d_in[B0 + 10], nullptr, h_b, T_TOK, DFF, DM, 1);
            gemm(h_b, w2b[L], (const float*)d_in[B0 + 12], tmp_f, nullptr, T_TOK, DM, DFF, 0);
            ln_residual_kernel<<<T_TOK / 8, 256, 0, stream>>>(
                cx_f, tmp_f, (const float*)d_in[B0 + 13], (const float*)d_in[B0 + 14], cx_b);
        }
    };

    // embeddings + Rasch
    embed_kernel<<<(unsigned)(TD / 256), 256, 0, stream>>>(
        (const float*)d_in[0], (const float*)d_in[1], (const float*)d_in[2],
        (const float*)d_in[3], (const float*)d_in[4], (const int*)d_in[101],
        (const int*)d_in[102], (const int*)d_in[103], qe_f, qe_b, y_f, y_b);

    // blocks_1 on y (self, mask_k=1, FFN)
    run_layer(0, 1, 1, y_f, y_b, y_b);
    run_layer(1, 1, 1, y_f, y_b, y_b);

    // x starts as q_embed_data
    hipMemcpyAsync(x_f, qe_f, TD * 4, hipMemcpyDeviceToDevice, stream);
    hipMemcpyAsync(x_b, qe_b, TD * 2, hipMemcpyDeviceToDevice, stream);

    // blocks_2: self(no-FFN) / cross-with-y(FFN), twice
    run_layer(2, 1, 0, x_f, x_b, x_b);
    run_layer(3, 0, 1, x_f, x_b, y_b);
    run_layer(4, 1, 0, x_f, x_b, x_b);
    run_layer(5, 0, 1, x_f, x_b, y_b);

    // head: concat -> 512 relu -> 256 relu -> 1 sigmoid
    concat_kernel<<<(unsigned)((size_t)T_TOK * 1024 / 256), 256, 0, stream>>>(x_b, qe_b,
                                                                              h_b);
    gemm(h_b, o1wb, (const float*)d_in[96], nullptr, qk_b, T_TOK, 512, 1024, 1);
    gemm(qk_b, o2wb, (const float*)d_in[98], tmp_f, nullptr, T_TOK, 256, 512, 1);
    out3_kernel<<<T_TOK / 8, 256, 0, stream>>>(tmp_f, (const float*)d_in[99],
                                               (const float*)d_in[100], (float*)d_out);
}
